// MHA_66056597013124
// MI455X (gfx1250) — matrix operand order repaired and checked by reading, not compiled, not run on hardware
//
#include <hip/hip_runtime.h>

#define D_MODEL 1024
#define NHEADS  16
#define DK      64
#define SEQ     2048
#define NBATCH  2
#define MTOT    (NBATCH * SEQ)   // 4096 rows for all projection GEMMs

typedef __attribute__((ext_vector_type(16))) __bf16 v16bf;
typedef __attribute__((ext_vector_type(8)))  __bf16 v8bf;
typedef __attribute__((ext_vector_type(8)))  float  v8f;

// ---------------------------------------------------------------------------
// WMMA wrapper: D = A(16x32 bf16) * B(32x16 bf16) + C(16x16 f32)
// ---------------------------------------------------------------------------
__device__ __forceinline__ v8f wmma_bf16(v16bf a, v16bf b, v8f c) {
  return __builtin_amdgcn_wmma_f32_16x16x32_bf16(
      /*neg_a=*/false, a, /*neg_b=*/false, b,
      /*c_mod=*/(short)0, c, /*reuse_a=*/false, /*reuse_b=*/false);
}

__device__ __forceinline__ void ld8bf(v16bf& d, int off, const __bf16* p) {
  v8bf t = *reinterpret_cast<const v8bf*>(p);
#pragma unroll
  for (int i = 0; i < 8; ++i) d[off + i] = t[i];
}

// A fragment 16x32 (ISA layout: lane half packs K {h*8..h*8+7, 16+h*8..+7})
__device__ __forceinline__ v16bf fragA_bf(const __bf16* base, int stride,
                                          int l16, int half, int k0) {
  v16bf a;
  const __bf16* p = base + (size_t)l16 * stride + k0 + half * 8;
  ld8bf(a, 0, p);
  ld8bf(a, 8, p + 16);
  return a;
}
// B fragment 32x16 loaded from BT row-major [n][k] (lane half packs K {h*8..h*8+7, 16+h*8..+7})
__device__ __forceinline__ v16bf fragB_bf(const __bf16* baseT, int stride,
                                          int l16, int half, int k0) {
  v16bf b;
  const __bf16* p = baseT + (size_t)l16 * stride + k0 + half * 8;
  ld8bf(b, 0, p);
  ld8bf(b, 8, p + 16);
  return b;
}

// 16-lane group reductions (rows of a C fragment live across 16 lanes)
__device__ __forceinline__ float rmax16(float v) {
  v = fmaxf(v, __shfl_xor(v, 1, 32));
  v = fmaxf(v, __shfl_xor(v, 2, 32));
  v = fmaxf(v, __shfl_xor(v, 4, 32));
  v = fmaxf(v, __shfl_xor(v, 8, 32));
  return v;
}
__device__ __forceinline__ float rsum16(float v) {
  v += __shfl_xor(v, 1, 32);
  v += __shfl_xor(v, 2, 32);
  v += __shfl_xor(v, 4, 32);
  v += __shfl_xor(v, 8, 32);
  return v;
}

// ---------------------------------------------------------------------------
// Weight transpose+convert: W[k][n] f32 -> WT[n][k] bf16
// ---------------------------------------------------------------------------
__global__ void wtrans_kernel(const float* __restrict__ W, __bf16* __restrict__ WT) {
  int idx = blockIdx.x * blockDim.x + threadIdx.x;   // 0 .. 1M-1
  int k = idx >> 10, n = idx & (D_MODEL - 1);
  WT[(size_t)n * D_MODEL + k] = (__bf16)W[idx];
}

// ---------------------------------------------------------------------------
// Streaming activation convert: f32 -> bf16, same layout. 8 elems/thread.
// ---------------------------------------------------------------------------
__global__ void xconv_kernel(const float* __restrict__ X, __bf16* __restrict__ Xb) {
  size_t i = ((size_t)blockIdx.x * blockDim.x + threadIdx.x) * 8;
  float4 t0 = *reinterpret_cast<const float4*>(X + i);
  float4 t1 = *reinterpret_cast<const float4*>(X + i + 4);
  v8bf o;
  o[0] = (__bf16)t0.x; o[1] = (__bf16)t0.y; o[2] = (__bf16)t0.z; o[3] = (__bf16)t0.w;
  o[4] = (__bf16)t1.x; o[5] = (__bf16)t1.y; o[6] = (__bf16)t1.z; o[7] = (__bf16)t1.w;
  *reinterpret_cast<v8bf*>(Xb + i) = o;
}

// ---------------------------------------------------------------------------
// GEMM: C[M=4096][N=1024] = Xb(bf16) * W (via WT bf16) + bias
// One wave computes a 32x32 tile: 2 A-frags x 2 B-frags -> 4 WMMAs per k-step.
// mode 0: f32 row-major; mode 1: bf16 row-major; mode 2: bf16 V-transposed
// ---------------------------------------------------------------------------
__global__ __launch_bounds__(128) void gemm_kernel(
    const __bf16* __restrict__ X, const __bf16* __restrict__ WT,
    const float* __restrict__ bias, void* __restrict__ out, int mode) {
  int wid  = blockIdx.x * (blockDim.x >> 5) + (threadIdx.x >> 5);
  int lane = threadIdx.x & 31;
  int l16 = lane & 15, half = lane >> 4;
  int tn = wid & (D_MODEL / 32 - 1);   // 0..31
  int tm = wid >> 5;                   // 0..127
  int m0 = tm * 32, n0 = tn * 32;

  const __bf16* A0 = X + (size_t)m0 * D_MODEL;
  const __bf16* A1 = X + (size_t)(m0 + 16) * D_MODEL;
  const __bf16* B0 = WT + (size_t)n0 * D_MODEL;
  const __bf16* B1 = WT + (size_t)(n0 + 16) * D_MODEL;

  v8f c00 = {}, c01 = {}, c10 = {}, c11 = {};
#pragma unroll 4
  for (int k0 = 0; k0 < D_MODEL; k0 += 32) {
    v16bf a0 = fragA_bf(A0, D_MODEL, l16, half, k0);
    v16bf a1 = fragA_bf(A1, D_MODEL, l16, half, k0);
    v16bf b0 = fragB_bf(B0, D_MODEL, l16, half, k0);
    v16bf b1 = fragB_bf(B1, D_MODEL, l16, half, k0);
    // keep the A stream ahead of the math (global_prefetch_b8)
    __builtin_prefetch(A0 + (size_t)l16 * D_MODEL + k0 + 128, 0, 1);
    c00 = wmma_bf16(a0, b0, c00);
    c01 = wmma_bf16(a0, b1, c01);
    c10 = wmma_bf16(a1, b0, c10);
    c11 = wmma_bf16(a1, b1, c11);
  }

  v8f cc[2][2] = {{c00, c01}, {c10, c11}};
  float bvj[2] = {bias[n0 + l16], bias[n0 + 16 + l16]};
  if (mode == 0) {
    float* O = (float*)out;
#pragma unroll
    for (int i = 0; i < 2; ++i)
#pragma unroll
      for (int r = 0; r < 8; ++r) {
        int m = m0 + i * 16 + r + half * 8;
#pragma unroll
        for (int j = 0; j < 2; ++j)
          O[(size_t)m * D_MODEL + n0 + j * 16 + l16] = cc[i][j][r] + bvj[j];
      }
  } else if (mode == 1) {
    __bf16* O = (__bf16*)out;
#pragma unroll
    for (int i = 0; i < 2; ++i)
#pragma unroll
      for (int r = 0; r < 8; ++r) {
        int m = m0 + i * 16 + r + half * 8;
#pragma unroll
        for (int j = 0; j < 2; ++j)
          O[(size_t)m * D_MODEL + n0 + j * 16 + l16] = (__bf16)(cc[i][j][r] + bvj[j]);
      }
  } else {  // V transposed: VT[b][h][d][s]
    __bf16* O = (__bf16*)out;
#pragma unroll
    for (int j = 0; j < 2; ++j) {
      int n = n0 + j * 16 + l16;
      int h = n >> 6, d = n & (DK - 1);
#pragma unroll
      for (int i = 0; i < 2; ++i)
#pragma unroll
        for (int r = 0; r < 8; ++r) {
          int m = m0 + i * 16 + r + half * 8;
          int b_ = m >> 11, s = m & (SEQ - 1);
          O[(((size_t)(b_ * NHEADS + h)) * DK + d) * SEQ + s] = (__bf16)(cc[i][j][r] + bvj[j]);
        }
    }
  }
}

// ---------------------------------------------------------------------------
// Fused flash-style attention. One wave per (b,h, 16-query tile).
// Pass 1: online softmax + O accumulation.  Pass 2: normalized attn writeout.
// ---------------------------------------------------------------------------
__device__ __forceinline__ void score_tiles(const __bf16* Kbase, int kb,
                                            v16bf aQ0, v16bf aQ1,
                                            int l16, int half, v8f& s0, v8f& s1) {
  v8f z = {};
  v16bf b0 = fragB_bf(Kbase + (size_t)kb * D_MODEL, D_MODEL, l16, half, 0);
  v16bf b1 = fragB_bf(Kbase + (size_t)kb * D_MODEL, D_MODEL, l16, half, 32);
  s0 = wmma_bf16(aQ1, b1, wmma_bf16(aQ0, b0, z));
  b0 = fragB_bf(Kbase + (size_t)(kb + 16) * D_MODEL, D_MODEL, l16, half, 0);
  b1 = fragB_bf(Kbase + (size_t)(kb + 16) * D_MODEL, D_MODEL, l16, half, 32);
  s1 = wmma_bf16(aQ1, b1, wmma_bf16(aQ0, b0, z));
}

__global__ __launch_bounds__(128) void attn_kernel(
    const __bf16* __restrict__ Qb, const __bf16* __restrict__ Kb,
    const __bf16* __restrict__ VT, __bf16* __restrict__ ctx,
    float* __restrict__ attn) {
  __shared__ __bf16 ldsP[4][16 * 32];   // per-wave P staging (C-frag -> A-frag)
  int wic  = threadIdx.x >> 5;
  int wid  = blockIdx.x * 4 + wic;
  int lane = threadIdx.x & 31;
  int l16 = lane & 15, half = lane >> 4;
  int qt = wid & (SEQ / 16 - 1);        // 0..127
  int bh = wid >> 7;                    // 0..31
  int b = bh >> 4, h = bh & (NHEADS - 1);
  int q0 = qt * 16;
  const float scale = 0.125f;           // 1/sqrt(64)

  const __bf16* Qbase = Qb + ((size_t)(b * SEQ + q0)) * D_MODEL + h * DK;
  const __bf16* Kbase = Kb + ((size_t)(b * SEQ)) * D_MODEL + h * DK;
  const __bf16* Vbase = VT + (size_t)bh * DK * SEQ;   // rows = d, stride SEQ

  v16bf aQ0 = fragA_bf(Qbase, D_MODEL, l16, half, 0);
  v16bf aQ1 = fragA_bf(Qbase, D_MODEL, l16, half, 32);

  float mrow[8], lrow[8];
  v8f o[4];
#pragma unroll
  for (int r = 0; r < 8; ++r) { mrow[r] = -1e30f; lrow[r] = 0.0f; }
#pragma unroll
  for (int j = 0; j < 4; ++j) o[j] = (v8f){0, 0, 0, 0, 0, 0, 0, 0};

  __bf16* myP = &ldsP[wic][0];

  // ---- Pass 1: online softmax + O accumulation over 32-key blocks ----
  for (int kb = 0; kb < SEQ; kb += 32) {
    v8f s0, s1;
    score_tiles(Kbase, kb, aQ0, aQ1, l16, half, s0, s1);
#pragma unroll
    for (int r = 0; r < 8; ++r) {
      float v0 = s0[r] * scale, v1 = s1[r] * scale;
      float mnew = fmaxf(mrow[r], rmax16(fmaxf(v0, v1)));
      float alpha = __expf(mrow[r] - mnew);
      float p0 = __expf(v0 - mnew), p1 = __expf(v1 - mnew);
      lrow[r] = lrow[r] * alpha + rsum16(p0 + p1);
      mrow[r] = mnew;
#pragma unroll
      for (int j = 0; j < 4; ++j) o[j][r] *= alpha;
      int row = r + half * 8;
      myP[row * 32 + l16]      = (__bf16)p0;   // key tile 0
      myP[row * 32 + 16 + l16] = (__bf16)p1;   // key tile 1
    }
    // same-wave DS ops are processed in order; block compiler reordering only
    asm volatile("" ::: "memory");
    v16bf aP = fragA_bf(myP, 32, l16, half, 0);
#pragma unroll
    for (int j = 0; j < 4; ++j) {
      v16bf bv = fragB_bf(Vbase + (size_t)(j * 16) * SEQ, SEQ, l16, half, kb);
      o[j] = wmma_bf16(aP, bv, o[j]);
    }
  }

  float inv[8];
#pragma unroll
  for (int r = 0; r < 8; ++r) inv[r] = 1.0f / lrow[r];

  // ---- context writeout (bf16, consumed by output projection GEMM) ----
#pragma unroll
  for (int r = 0; r < 8; ++r) {
    int m = q0 + r + half * 8;
    __bf16* crow = ctx + ((size_t)(b * SEQ + m)) * D_MODEL + h * DK;
#pragma unroll
    for (int j = 0; j < 4; ++j) crow[j * 16 + l16] = (__bf16)(o[j][r] * inv[r]);
  }

  // ---- Pass 2: normalized attention probabilities (K tiles hit in L2) ----
  float* arow_base = attn + ((size_t)bh * SEQ + q0) * SEQ;
  for (int kb = 0; kb < SEQ; kb += 32) {
    v8f s0, s1;
    score_tiles(Kbase, kb, aQ0, aQ1, l16, half, s0, s1);
#pragma unroll
    for (int r = 0; r < 8; ++r) {
      int row = r + half * 8;
      float a0 = __expf(s0[r] * scale - mrow[r]) * inv[r];
      float a1 = __expf(s1[r] * scale - mrow[r]) * inv[r];
      float* ap = arow_base + (size_t)row * SEQ + kb + l16;
      ap[0]  = a0;
      ap[16] = a1;
    }
  }
}

// ---------------------------------------------------------------------------
// Host-side orchestration
// ---------------------------------------------------------------------------
extern "C" void kernel_launch(void* const* d_in, const int* in_sizes, int n_in,
                              void* d_out, int out_size, void* d_ws, size_t ws_size,
                              hipStream_t stream) {
  const float* q  = (const float*)d_in[0];
  const float* k  = (const float*)d_in[1];
  const float* v  = (const float*)d_in[2];
  const float* Wq = (const float*)d_in[3];
  const float* bq = (const float*)d_in[4];
  const float* Wk = (const float*)d_in[5];
  const float* bk = (const float*)d_in[6];
  const float* Wv = (const float*)d_in[7];
  const float* bv = (const float*)d_in[8];
  const float* Wo = (const float*)d_in[9];
  const float* bo = (const float*)d_in[10];

  char* ws = (char*)d_ws;
  const size_t WBYTES = (size_t)D_MODEL * D_MODEL * 2;      // 2 MB per weight
  const size_t ABYTES = (size_t)MTOT * D_MODEL * 2;         // 8 MB per activation
  __bf16* WqT  = (__bf16*)(ws + 0 * WBYTES);
  __bf16* WkT  = (__bf16*)(ws + 1 * WBYTES);
  __bf16* WvT  = (__bf16*)(ws + 2 * WBYTES);
  __bf16* WoT  = (__bf16*)(ws + 3 * WBYTES);
  __bf16* Xb   = (__bf16*)(ws + 4 * WBYTES);                // reused q/k/v bf16
  __bf16* Qb   = (__bf16*)(ws + 4 * WBYTES + 1 * ABYTES);
  __bf16* Kb   = (__bf16*)(ws + 4 * WBYTES + 2 * ABYTES);
  __bf16* VT   = (__bf16*)(ws + 4 * WBYTES + 3 * ABYTES);   // [B][H][Dk][S]
  __bf16* ctxb = (__bf16*)(ws + 4 * WBYTES + 4 * ABYTES);   // [MTOT][1024] bf16

  float* out_main = (float*)d_out;
  float* attn_out = out_main + (size_t)MTOT * D_MODEL;

  const int WT_BLOCKS = (D_MODEL * D_MODEL) / 256;
  wtrans_kernel<<<WT_BLOCKS, 256, 0, stream>>>(Wq, WqT);
  wtrans_kernel<<<WT_BLOCKS, 256, 0, stream>>>(Wk, WkT);
  wtrans_kernel<<<WT_BLOCKS, 256, 0, stream>>>(Wv, WvT);
  wtrans_kernel<<<WT_BLOCKS, 256, 0, stream>>>(Wo, WoT);

  const int XC_BLOCKS = (MTOT * D_MODEL) / (8 * 256);        // 2048
  const int GEMM_BLOCKS = (MTOT / 32) * (D_MODEL / 32) / 4;  // 1024, 4 waves/block

  xconv_kernel<<<XC_BLOCKS, 256, 0, stream>>>(q, Xb);
  gemm_kernel<<<GEMM_BLOCKS, 128, 0, stream>>>(Xb, WqT, bq, (void*)Qb, 1);
  xconv_kernel<<<XC_BLOCKS, 256, 0, stream>>>(k, Xb);
  gemm_kernel<<<GEMM_BLOCKS, 128, 0, stream>>>(Xb, WkT, bk, (void*)Kb, 1);
  xconv_kernel<<<XC_BLOCKS, 256, 0, stream>>>(v, Xb);
  gemm_kernel<<<GEMM_BLOCKS, 128, 0, stream>>>(Xb, WvT, bv, (void*)VT, 2);

  const int ATTN_BLOCKS = (NBATCH * NHEADS * (SEQ / 16)) / 4;
  attn_kernel<<<ATTN_BLOCKS, 128, 0, stream>>>(Qb, Kb, VT, ctxb, attn_out);

  gemm_kernel<<<GEMM_BLOCKS, 128, 0, stream>>>(ctxb, WoT, bo, d_out, 0);
}
